// MultiHeadAdjAttentionLayer_11166914970284
// MI455X (gfx1250) — matrix operand order repaired and checked by reading, not compiled, not run on hardware
//
#include <hip/hip_runtime.h>
#include <hip/hip_bf16.h>

// ---------------------------------------------------------------------------
// MultiHeadAdjAttentionLayer for MI455X (gfx1250, wave32, WMMA).
//
// All matmuls run through v_wmma_f32_16x16x32_f16 (f32 accumulate). Global
// data stays f32 (reference dtype); tiles are converted to f16 while staging
// into LDS with float4 global loads. LDS tiles are stored "fragment-ready"
// (k contiguous per row) so every WMMA operand fragment is built from
// 16B-aligned ds_load_b128s + register shuffles (no half-packing VALU ops).
// The (B,NH,S,S) attention region of d_out doubles as scratch for raw masked
// scores -> in-place softmax -> probs, saving 512MB of workspace.
// ---------------------------------------------------------------------------

typedef __attribute__((ext_vector_type(4)))  _Float16 v4h;
typedef __attribute__((ext_vector_type(8)))  _Float16 v8h;
typedef __attribute__((ext_vector_type(16))) _Float16 v16h;
typedef __attribute__((ext_vector_type(4)))  float    v4f;
typedef __attribute__((ext_vector_type(8)))  float    v8f;

#define NEGV (-10000000000.0f)

__device__ __forceinline__ v16h cat8(v8h lo, v8h hi)
{
    return __builtin_shufflevector(lo, hi, 0, 1, 2, 3, 4, 5, 6, 7,
                                   8, 9, 10, 11, 12, 13, 14, 15);
}

// Generic batched WMMA GEMM:
//   C[z] = epilogue( A[z] (MxK) x B[z] (KxN, or NxK if TRANSB) )
//   z -> (b = z/nh, h = z%nh);  ptr += b*Sb + h*Sh for A/B/C.
// Epilogue: c = (mask0[n]==0 || mask1[n]==0 ? NEGV : c) * scale + bias[n]
// 256 threads = 8 waves arranged 4x2; wave tile = (BM/4) x (BN/2).
template <int BM, int BN, bool TRANSB>
__global__ __launch_bounds__(256) void gemm_wmma(
    const float* __restrict__ A, int lda, long aSb, long aSh,
    const float* __restrict__ Bm, int ldb, long bSb, long bSh,
    float* __restrict__ C, int ldc, long cSb, long cSh,
    const float* __restrict__ bias,
    const int* __restrict__ mask0, const int* __restrict__ mask1,
    float scale, int nh, int M, int N, int K)
{
    constexpr int BK  = 32;         // one WMMA K-step per tile
    constexpr int LDK = BK + 8;     // LDS row stride (halfs); 80B keeps 16B align
    constexpr int WM  = BM / 4;     // wave tile rows (4 waves in M)
    constexpr int WN  = BN / 2;     // wave tile cols (2 waves in N)
    constexpr int FM  = WM / 16;    // 16x16 fragments per wave (M)
    constexpr int FN  = WN / 16;    // 16x16 fragments per wave (N)

    // Both tiles stored row-major over K (fragment-ready).
    __shared__ _Float16 As[BM * LDK];   // As[m][k]
    __shared__ _Float16 Bs[BN * LDK];   // Bs[n][k]

    const int tid   = threadIdx.x;
    const int lane  = tid & 31;
    const int wid   = tid >> 5;
    const int waveM = wid >> 1;     // 0..3
    const int waveN = wid & 1;      // 0..1

    const int z = blockIdx.z;
    const int b = z / nh;
    const int h = z % nh;
    const float* Ap = A  + (long)b * aSb + (long)h * aSh;
    const float* Bp = Bm + (long)b * bSb + (long)h * bSh;
    float*       Cp = C  + (long)b * cSb + (long)h * cSh;
    const int* m0p = mask0 ? mask0 + (long)b * N : nullptr;
    const int* m1p = mask1 ? mask1 + (long)b * N : nullptr;

    const int m0 = blockIdx.y * BM;
    const int n0 = blockIdx.x * BN;

    v8f acc[FM][FN];
#pragma unroll
    for (int fm = 0; fm < FM; ++fm)
#pragma unroll
        for (int fn = 0; fn < FN; ++fn)
            acc[fm][fn] = {};

    const int am    = waveM * WM;         // wave offset inside block tile
    const int an    = waveN * WN;
    const int mlane = lane & 15;
    const int kqA   = (lane & 16) >> 1;   // 0 or 8  (A frag K quarter)
    const int kbB   = (lane & 16) >> 1;   // 0 or 8  (B frag K quarter)

    for (int kk = 0; kk < K; kk += BK) {
        // --- prefetch next k-tile (global_prefetch_b8) ---
        if (kk + BK < K) {
            __builtin_prefetch(&Ap[(long)(m0 + (tid & (BM - 1))) * lda + kk + BK], 0, 1);
            if (TRANSB)
                __builtin_prefetch(&Bp[(long)(n0 + (tid & (BN - 1))) * ldb + kk + BK], 0, 1);
            else
                __builtin_prefetch(&Bp[(long)(kk + BK + (tid & (BK - 1))) * ldb + n0], 0, 1);
        }

        // --- stage A tile (BM x BK): float4 loads, packed f16 stores ---
#pragma unroll
        for (int idx = tid; idx < BM * (BK / 4); idx += 256) {
            const int r  = idx >> 3;          // BK/4 == 8
            const int c4 = idx & 7;
            v4f x = *(const v4f*)&Ap[(long)(m0 + r) * lda + kk + c4 * 4];
            v4h hx = { (_Float16)x[0], (_Float16)x[1],
                       (_Float16)x[2], (_Float16)x[3] };
            *(v4h*)&As[r * LDK + c4 * 4] = hx;
        }
        // --- stage B tile into Bs[n][k] ---
        if (TRANSB) {   // global B is (N x K): rows already k-contiguous
#pragma unroll
            for (int idx = tid; idx < BN * (BK / 4); idx += 256) {
                const int n  = idx >> 3;
                const int c4 = idx & 7;
                v4f x = *(const v4f*)&Bp[(long)(n0 + n) * ldb + kk + c4 * 4];
                v4h hx = { (_Float16)x[0], (_Float16)x[1],
                           (_Float16)x[2], (_Float16)x[3] };
                *(v4h*)&Bs[n * LDK + c4 * 4] = hx;
            }
        } else {        // global B is (K x N): transpose while staging
#pragma unroll
            for (int idx = tid; idx < BK * (BN / 4); idx += 256) {
                const int k2 = idx / (BN / 4);
                const int n4 = idx % (BN / 4);
                v4f x = *(const v4f*)&Bp[(long)(kk + k2) * ldb + n0 + n4 * 4];
#pragma unroll
                for (int j = 0; j < 4; ++j)
                    Bs[(n4 * 4 + j) * LDK + k2] = (_Float16)x[j];
            }
        }
        __syncthreads();

        // --- build fragments (ISA 7.12.2 layouts) from contiguous LDS ---
        v16h afrag[FM], bfrag[FN];
#pragma unroll
        for (int fm = 0; fm < FM; ++fm) {
            const _Float16* p = &As[(am + fm * 16 + mlane) * LDK];
            v8h lo = *(const v8h*)(p + kqA);        // K = kqA .. kqA+7
            v8h hi = *(const v8h*)(p + 16 + kqA);   // K = 16+kqA .. +7
            afrag[fm] = cat8(lo, hi);
        }
#pragma unroll
        for (int fn = 0; fn < FN; ++fn) {
            const _Float16* p = &Bs[(an + fn * 16 + mlane) * LDK + kbB];
            v8h lo = *(const v8h*)(p);              // K = kbB .. kbB+7
            v8h hi = *(const v8h*)(p + 16);         // K = 16+kbB .. +7
            bfrag[fn] = cat8(lo, hi);
        }

        // --- v_wmma_f32_16x16x32_f16 ---
#pragma unroll
        for (int fm = 0; fm < FM; ++fm)
#pragma unroll
            for (int fn = 0; fn < FN; ++fn)
                acc[fm][fn] = __builtin_amdgcn_wmma_f32_16x16x32_f16(
                    false, afrag[fm], false, bfrag[fn],
                    (short)0, acc[fm][fn], false, false);
        __syncthreads();
    }

    // --- epilogue: mask / scale / bias, store f32 (coalesced across lanes) ---
    const int mb = (lane & 16) >> 1;  // 0 or 8 (C frag row half)
#pragma unroll
    for (int fn = 0; fn < FN; ++fn) {
        const int n = n0 + an + fn * 16 + (lane & 15);
        const float bv = bias ? bias[n] : 0.0f;
        const bool dead = (m0p && m0p[n] == 0) || (m1p && m1p[n] == 0);
#pragma unroll
        for (int fm = 0; fm < FM; ++fm) {
            const int mr = m0 + am + fm * 16 + mb;
#pragma unroll
            for (int v = 0; v < 8; ++v) {
                float c = acc[fm][fn][v];
                if (dead) c = NEGV;
                c = c * scale + bv;
                Cp[(long)(mr + v) * ldc + n] = c;
            }
        }
    }
}

// In-place row softmax, row length 1024, 256 threads (4 elems/thread).
__global__ __launch_bounds__(256) void softmax_rows(float* __restrict__ data)
{
    const long row = blockIdx.x;
    float* p = data + row * 1024L;
    const int tid = threadIdx.x;
    __shared__ float red[256];

    float vals[4];
    float m = -3.4e38f;
#pragma unroll
    for (int i = 0; i < 4; ++i) {
        vals[i] = p[tid + i * 256];
        m = fmaxf(m, vals[i]);
    }
    red[tid] = m;
    __syncthreads();
    for (int s = 128; s > 0; s >>= 1) {
        if (tid < s) red[tid] = fmaxf(red[tid], red[tid + s]);
        __syncthreads();
    }
    m = red[0];
    __syncthreads();

    float sum = 0.0f;
#pragma unroll
    for (int i = 0; i < 4; ++i) {
        vals[i] = __expf(vals[i] - m);
        sum += vals[i];
    }
    red[tid] = sum;
    __syncthreads();
    for (int s = 128; s > 0; s >>= 1) {
        if (tid < s) red[tid] += red[tid + s];
        __syncthreads();
    }
    const float inv = 1.0f / red[0];
#pragma unroll
    for (int i = 0; i < 4; ++i)
        p[tid + i * 256] = vals[i] * inv;
}

extern "C" void kernel_launch(void* const* d_in, const int* in_sizes, int n_in,
                              void* d_out, int out_size, void* d_ws, size_t ws_size,
                              hipStream_t stream)
{
    // dims
    constexpr long S = 1024, HID = 1024, NH = 16, HD = 64, STR = 128;
    constexpr long MS = 8 * S;                // B*S = 8192 rows
    constexpr long CATW = HID + STR;          // 1152

    const float* query    = (const float*)d_in[0];
    const float* key      = (const float*)d_in[1];
    const float* value    = (const float*)d_in[2];
    const float* x_rsa    = (const float*)d_in[3];
    const int*   mask     = (const int*)d_in[4];
    const int*   rsa_mask = (const int*)d_in[5];
    const float* Wq  = (const float*)d_in[6];
    const float* bq  = (const float*)d_in[7];
    const float* Wk  = (const float*)d_in[8];
    const float* bk  = (const float*)d_in[9];
    const float* Wv  = (const float*)d_in[10];
    const float* bv  = (const float*)d_in[11];
    const float* Wsk = (const float*)d_in[12];
    const float* bsk = (const float*)d_in[13];
    const float* Wsv = (const float*)d_in[14];
    const float* bsv = (const float*)d_in[15];
    const float* Wo  = (const float*)d_in[16];
    const float* bo  = (const float*)d_in[17];

    float* out = (float*)d_out;
    float* ws  = (float*)d_ws;

    // workspace layout (f32)
    float* Qb  = ws;                    // 8192x1024
    float* Kb  = Qb  + MS * HID;
    float* Vb  = Kb  + MS * HID;
    float* sKb = Vb  + MS * HID;        // 8192x128
    float* sVb = sKb + MS * STR;
    float* cat = sVb + MS * STR;        // 8192x1152 concat(x, rsa_x)

    // d_out layout: [out 8M][attention 128M][rsa_attention 8M]
    float* attn = out + 8L * S * HID;                    // 8,388,608
    float* rsa  = attn + 8L * NH * S * S;                // +134,217,728

    const dim3 blk(256);

    // ---- projections: X @ W + b (WMMA) ----
    gemm_wmma<128,128,false><<<dim3(8,64,1), blk, 0, stream>>>(
        query, HID, 0, 0, Wq, HID, 0, 0, Qb, HID, 0, 0,
        bq, nullptr, nullptr, 1.0f, 1, MS, HID, HID);
    gemm_wmma<128,128,false><<<dim3(8,64,1), blk, 0, stream>>>(
        key,   HID, 0, 0, Wk, HID, 0, 0, Kb, HID, 0, 0,
        bk, nullptr, nullptr, 1.0f, 1, MS, HID, HID);
    gemm_wmma<128,128,false><<<dim3(8,64,1), blk, 0, stream>>>(
        value, HID, 0, 0, Wv, HID, 0, 0, Vb, HID, 0, 0,
        bv, nullptr, nullptr, 1.0f, 1, MS, HID, HID);
    gemm_wmma<128,128,false><<<dim3(1,64,1), blk, 0, stream>>>(
        key,   HID, 0, 0, Wsk, STR, 0, 0, sKb, STR, 0, 0,
        bsk, nullptr, nullptr, 1.0f, 1, MS, STR, HID);
    gemm_wmma<128,128,false><<<dim3(1,64,1), blk, 0, stream>>>(
        value, HID, 0, 0, Wsv, STR, 0, 0, sVb, STR, 0, 0,
        bsv, nullptr, nullptr, 1.0f, 1, MS, STR, HID);

    // ---- scores: Q K^T, mask, /sqrt(HD); written into d_out attn region ----
    gemm_wmma<128,128,true><<<dim3(8,8,128), blk, 0, stream>>>(
        Qb, HID, S*HID, HD,  Kb, HID, S*HID, HD,
        attn, S, NH*S*S, S*S,
        nullptr, mask, nullptr, 0.125f, (int)NH, S, S, HD);

    // ---- rsa scores: x_rsa strK^T, both masks, no scale ----
    gemm_wmma<128,128,true><<<dim3(8,8,8), blk, 0, stream>>>(
        x_rsa, STR, S*STR, 0,  sKb, STR, S*STR, 0,
        rsa, S, S*S, 0,
        nullptr, mask, rsa_mask, 1.0f, 1, S, S, STR);

    // ---- softmax in place over attn + rsa (contiguous): 139264 rows ----
    softmax_rows<<<dim3((8*NH + 8) * S), blk, 0, stream>>>(attn);

    // ---- x = attn @ V -> concat buffer cols [0,1024) ----
    gemm_wmma<128,64,false><<<dim3(1,8,128), blk, 0, stream>>>(
        attn, S, NH*S*S, S*S,  Vb, HID, S*HID, HD,
        cat, CATW, S*CATW, HD,
        nullptr, nullptr, nullptr, 1.0f, (int)NH, S, HD, S);

    // ---- rsa_x = rsa_attn @ strV -> concat cols [1024,1152) ----
    gemm_wmma<128,128,false><<<dim3(1,8,8), blk, 0, stream>>>(
        rsa, S, S*S, 0,  sVb, STR, S*STR, 0,
        cat + HID, CATW, S*CATW, 0,
        nullptr, nullptr, nullptr, 1.0f, 1, S, STR, S);

    // ---- out = concat @ Wo + bo ----
    gemm_wmma<128,128,false><<<dim3(8,64,1), blk, 0, stream>>>(
        cat, CATW, 0, 0,  Wo, HID, 0, 0,  out, HID, 0, 0,
        bo, nullptr, nullptr, 1.0f, 1, MS, HID, CATW);
}
